// MoEBlock_71708773974440
// MI455X (gfx1250) — matrix operand order repaired and checked by reading, not compiled, not run on hardware
//
#include <hip/hip_runtime.h>
#include <hip/hip_bf16.h>
#include <math.h>

// ---- problem dims (match reference) ----
#define B_    2
#define T_    1024
#define C_    1024
#define HQ_   16
#define HKV_  4
#define D_    64
#define E_    8
#define H_    2752
#define N_    (B_*T_)          // 2048 tokens
#define CAP_  640              // int(1.25 * N * 2 / E)

typedef __attribute__((ext_vector_type(16))) _Float16 v16h;
typedef __attribute__((ext_vector_type(8)))  float    v8f;
typedef __attribute__((ext_vector_type(8)))  _Float16 h8v;
typedef __attribute__((ext_vector_type(4)))  _Float16 h4v;
typedef __attribute__((ext_vector_type(4)))  float    f4v;

union AFrag { v16h v; h8v h[2]; _Float16 e[16]; };
union CFrag { v8f  v; float  e[8]; };

// ---------------------------------------------------------------------------
// WMMA f16 GEMM:  C[M,N] = A[M,K] * Bt[N,K]^T   (A, Bt are f16, K-major)
// Epilogue: val = acc; if(rowScale) val *= rowScale[m*rsStride];
//           if(addend) val += addend[m*N+n]; accumulate ? C+= : C=
// Requires M%128==0, N%64==0, K%32==0 (true for all call sites here).
// ---------------------------------------------------------------------------
#define BM 128
#define BN 64
#define BK 32

__global__ __launch_bounds__(256)
void gemm_wmma_f16(const _Float16* __restrict__ A, const _Float16* __restrict__ Bt,
                   float* __restrict__ Cm, int M, int N, int K,
                   const float* __restrict__ rowScale, int rsStride,
                   const float* __restrict__ addend, int accumulate)
{
    __shared__ _Float16 sA[BM * BK];   // [row][k]  K contiguous
    __shared__ _Float16 sB[BN * BK];   // [col][k]  K contiguous

    const int tid  = threadIdx.x;
    const int lane = tid & 31;
    const int wid  = tid >> 5;            // 8 waves
    const int wm   = (wid >> 1) * 32;     // 0,32,64,96
    const int wn   = (wid & 1)  * 32;     // 0,32
    const int m0   = blockIdx.y * BM;
    const int n0   = blockIdx.x * BN;

    CFrag acc[2][2];
#pragma unroll
    for (int i = 0; i < 2; ++i)
#pragma unroll
        for (int j = 0; j < 2; ++j)
#pragma unroll
            for (int r = 0; r < 8; ++r) acc[i][j].e[r] = 0.f;

    const int ar  = lane & 15;            // A: row within 16-row tile
    const int ah  = (lane >> 4) * 8;      // A: K sub-base (0 or 8)
    const int bc  = lane & 15;            // B: col within 16-col tile
    const int bk2 = (lane >> 4) * 8;      // B: K sub-base (0 or 8)

    for (int k0 = 0; k0 < K; k0 += BK) {
#pragma unroll
        for (int i = 0; i < 2; ++i) {
            int chunk = i * 256 + tid;
            int row = chunk >> 2, co = (chunk & 3) * 8;
            *(h8v*)&sA[row * BK + co] =
                *(const h8v*)&A[(size_t)(m0 + row) * K + k0 + co];
        }
        {
            int col = tid >> 2, co = (tid & 3) * 8;
            *(h8v*)&sB[col * BK + co] =
                *(const h8v*)&Bt[(size_t)(n0 + col) * K + k0 + co];
        }
        if (k0 + BK < K) {
            __builtin_prefetch(&A[(size_t)(m0 + (tid >> 1)) * K + k0 + BK], 0, 1);
            __builtin_prefetch(&Bt[(size_t)(n0 + (tid >> 2)) * K + k0 + BK], 0, 1);
        }
        __syncthreads();

        AFrag a[2], b[2];
#pragma unroll
        for (int i = 0; i < 2; ++i) {
            const _Float16* ap = &sA[(wm + i * 16 + ar) * BK];
            a[i].h[0] = *(const h8v*)&ap[ah];
            a[i].h[1] = *(const h8v*)&ap[16 + ah];
        }
#pragma unroll
        for (int j = 0; j < 2; ++j) {
            const _Float16* bp = &sB[(wn + j * 16 + bc) * BK + bk2];
            b[j].h[0] = *(const h8v*)&bp[0];
            b[j].h[1] = *(const h8v*)&bp[16];
        }
#pragma unroll
        for (int i = 0; i < 2; ++i)
#pragma unroll
            for (int j = 0; j < 2; ++j)
                acc[i][j].v = __builtin_amdgcn_wmma_f32_16x16x32_f16(
                    false, a[i].v, false, b[j].v,
                    (short)0, acc[i][j].v, false, false);
        __syncthreads();
    }

    const int cr = lane & 15;
    const int rb = (lane >> 4) * 8;
#pragma unroll
    for (int i = 0; i < 2; ++i)
#pragma unroll
        for (int j = 0; j < 2; ++j)
#pragma unroll
            for (int r = 0; r < 8; ++r) {
                int gm = m0 + wm + i * 16 + rb + r;
                int gn = n0 + wn + j * 16 + cr;
                float val = acc[i][j].e[r];
                if (rowScale) val *= rowScale[(size_t)gm * rsStride];
                if (addend)   val += addend[(size_t)gm * N + gn];
                size_t o = (size_t)gm * N + gn;
                if (accumulate) Cm[o] += val; else Cm[o] = val;
            }
}

// ---------------------------------------------------------------------------
// Fused transpose + fp32->f16 convert: src[K][N] (fp32) -> dst[N][K] (f16)
// ---------------------------------------------------------------------------
__global__ __launch_bounds__(256)
void transpose_f32_to_f16(const float* __restrict__ src, _Float16* __restrict__ dst,
                          int K, int N)
{
    __shared__ float tile[32][33];
    const int kb = blockIdx.y * 32, nb = blockIdx.x * 32;
    const int tx = threadIdx.x, ty = threadIdx.y;
#pragma unroll
    for (int i = ty; i < 32; i += 8)
        tile[i][tx] = src[(size_t)(kb + i) * N + nb + tx];
    __syncthreads();
#pragma unroll
    for (int i = ty; i < 32; i += 8)
        dst[(size_t)(nb + i) * K + kb + tx] = (_Float16)tile[tx][i];
}

// ---------------------------------------------------------------------------
// RMSNorm: one block per row, fp32 in, f16 out (feeds WMMA GEMMs)
// ---------------------------------------------------------------------------
__global__ __launch_bounds__(256)
void rmsnorm_f16_kernel(const float* __restrict__ x, const float* __restrict__ w,
                        _Float16* __restrict__ out)
{
    __shared__ float sp[8];
    const int row = blockIdx.x;
    const int tid = threadIdx.x, lane = tid & 31, wid = tid >> 5;
    const float* xr = x + (size_t)row * C_;
    float s = 0.f;
    for (int c = tid; c < C_; c += 256) { float v = xr[c]; s += v * v; }
#pragma unroll
    for (int o = 16; o > 0; o >>= 1) s += __shfl_xor(s, o, 32);
    if (lane == 0) sp[wid] = s;
    __syncthreads();
    if (tid == 0) {
        float t = 0.f;
        for (int i = 0; i < 8; ++i) t += sp[i];
        sp[0] = rsqrtf(t / (float)C_ + 1e-6f);
    }
    __syncthreads();
    const float sc = sp[0];
    for (int c = tid; c < C_; c += 256)
        out[(size_t)row * C_ + c] = (_Float16)(xr[c] * sc * w[c]);
}

// ---------------------------------------------------------------------------
// RoPE (in place, fp32) on [N, nH, D] with cache [T, D/2, 2]
// ---------------------------------------------------------------------------
__global__ void rope_kernel(float* __restrict__ x, const float* __restrict__ cache, int nH)
{
    int idx = blockIdx.x * blockDim.x + threadIdx.x;
    int tot = N_ * nH * (D_ / 2);
    if (idx >= tot) return;
    int p = idx % (D_ / 2);
    int h = (idx / (D_ / 2)) % nH;
    int n = idx / ((D_ / 2) * nH);
    int t = n % T_;
    float c = cache[((size_t)t * (D_ / 2) + p) * 2 + 0];
    float s = cache[((size_t)t * (D_ / 2) + p) * 2 + 1];
    size_t base = ((size_t)n * nH + h) * D_ + 2 * p;
    float x0 = x[base], x1 = x[base + 1];
    x[base]     = x0 * c - x1 * s;
    x[base + 1] = x0 * s + x1 * c;
}

// ---------------------------------------------------------------------------
// WMMA flash attention (causal, GQA). Block = 4 waves = 64 query rows of one
// (b,hq). Each wave owns 16 query rows. Key tiles of 32. Q,K staged [row][d]
// (K-dim contiguous = B^T layout for QK^T); V staged transposed [d][s] so PV
// B-fragments are contiguous; P redistributed C->A layout via per-wave LDS.
// ---------------------------------------------------------------------------
__global__ __launch_bounds__(128)
void flash_attn_wmma(const float* __restrict__ q, const float* __restrict__ k,
                     const float* __restrict__ v, _Float16* __restrict__ y)
{
    __shared__ _Float16 sQ[64 * 64];     // [qrow][d]
    __shared__ _Float16 sK[32 * 64];     // [s][d]
    __shared__ _Float16 sVt[64 * 32];    // [d][s]
    __shared__ _Float16 sP[4 * 16 * 32]; // per-wave P tile [qrow][s]

    const int q0 = blockIdx.x * 64;
    const int bh = blockIdx.y;
    const int b = bh / HQ_, hq = bh % HQ_, hk = hq / (HQ_ / HKV_);
    const int tid = threadIdx.x, lane = tid & 31, wid = tid >> 5;

    const int ar  = lane & 15;
    const int ah  = (lane >> 4) * 8;
    const int bc  = lane & 15;
    const int bk2 = (lane >> 4) * 8;
    const int cr  = lane & 15;
    const int rb  = (lane >> 4) * 8;
    const int qbase = q0 + wid * 16;     // this wave's first query row

    // stage Q tile 64x64 (fp32 -> f16), 1024 f4v chunks, 8 per thread
#pragma unroll
    for (int i = 0; i < 8; ++i) {
        int chunk = i * 128 + tid;
        int qr = chunk >> 4, d = (chunk & 15) * 4;
        f4v f = *(const f4v*)&q[((size_t)(b * T_ + q0 + qr) * HQ_ + hq) * D_ + d];
        h4v h; h.x = (_Float16)f.x; h.y = (_Float16)f.y;
        h.z = (_Float16)f.z; h.w = (_Float16)f.w;
        *(h4v*)&sQ[qr * D_ + d] = h;
    }
    __syncthreads();

    // per-wave Q fragments (16 x 64 = 2 A-frags), kept in registers
    AFrag aq[2];
    {
        const _Float16* ap = &sQ[(wid * 16 + ar) * D_];
        aq[0].h[0] = *(const h8v*)&ap[ah];
        aq[0].h[1] = *(const h8v*)&ap[16 + ah];
        aq[1].h[0] = *(const h8v*)&ap[32 + ah];
        aq[1].h[1] = *(const h8v*)&ap[48 + ah];
    }

    CFrag o[4];
#pragma unroll
    for (int j = 0; j < 4; ++j)
#pragma unroll
        for (int r = 0; r < 8; ++r) o[j].e[r] = 0.f;
    float mrow[8], lrow[8];
#pragma unroll
    for (int r = 0; r < 8; ++r) { mrow[r] = -3.0e38f; lrow[r] = 0.f; }

    const int numTiles = q0 / 32 + 2;    // keys 0 .. q0+63
    for (int tIdx = 0; tIdx < numTiles; ++tIdx) {
        const int s0 = tIdx * 32;
        // stage K tile [s][d] (fp32 -> f16): 512 f4v chunks, 4 per thread
#pragma unroll
        for (int i = 0; i < 4; ++i) {
            int chunk = i * 128 + tid;
            int s = chunk >> 4, d = (chunk & 15) * 4;
            f4v f = *(const f4v*)&k[((size_t)(b * T_ + s0 + s) * HKV_ + hk) * D_ + d];
            h4v h; h.x = (_Float16)f.x; h.y = (_Float16)f.y;
            h.z = (_Float16)f.z; h.w = (_Float16)f.w;
            *(h4v*)&sK[s * D_ + d] = h;
        }
        // stage V transposed [d][s]: 2048 scalars, 16 per thread
#pragma unroll
        for (int i = 0; i < 16; ++i) {
            int idx = i * 128 + tid;
            int s = idx >> 6, d = idx & 63;
            sVt[d * 32 + s] =
                (_Float16)v[((size_t)(b * T_ + s0 + s) * HKV_ + hk) * D_ + d];
        }
        __syncthreads();

        if (s0 <= qbase + 15) {          // wave-uniform: tile not fully masked
            // ---- S = Q @ K^T : 16x32 scores as two 16x16 C-frags ----
            CFrag sf[2];
#pragma unroll
            for (int sub = 0; sub < 2; ++sub) {
                AFrag bk0, bk1;
                const _Float16* bp = &sK[(sub * 16 + bc) * D_];
                bk0.h[0] = *(const h8v*)&bp[bk2];
                bk0.h[1] = *(const h8v*)&bp[bk2 + 16];
                bk1.h[0] = *(const h8v*)&bp[32 + bk2];
                bk1.h[1] = *(const h8v*)&bp[32 + bk2 + 16];
                CFrag z;
#pragma unroll
                for (int r = 0; r < 8; ++r) z.e[r] = 0.f;
                z.v = __builtin_amdgcn_wmma_f32_16x16x32_f16(
                    false, aq[0].v, false, bk0.v, (short)0, z.v, false, false);
                z.v = __builtin_amdgcn_wmma_f32_16x16x32_f16(
                    false, aq[1].v, false, bk1.v, (short)0, z.v, false, false);
                sf[sub] = z;
            }
            // ---- scale + causal mask + online softmax ----
            const bool needMask = (s0 + 31 > qbase);
            float alphar[8];
#pragma unroll
            for (int r = 0; r < 8; ++r) {
                const int qrow = qbase + rb + r;
#pragma unroll
                for (int sub = 0; sub < 2; ++sub) {
                    float sv = sf[sub].e[r] * 0.125f;     // 1/sqrt(64)
                    if (needMask && (s0 + sub * 16 + cr > qrow)) sv = -3.0e38f;
                    sf[sub].e[r] = sv;
                }
                float vmax = fmaxf(sf[0].e[r], sf[1].e[r]);
#pragma unroll
                for (int os = 8; os > 0; os >>= 1)
                    vmax = fmaxf(vmax, __shfl_xor(vmax, os, 16));
                float mn = fmaxf(mrow[r], vmax);
                alphar[r] = __expf(mrow[r] - mn);
                mrow[r] = mn;
                sf[0].e[r] = __expf(sf[0].e[r] - mn);
                sf[1].e[r] = __expf(sf[1].e[r] - mn);
                float rs = sf[0].e[r] + sf[1].e[r];
#pragma unroll
                for (int os = 8; os > 0; os >>= 1)
                    rs += __shfl_xor(rs, os, 16);
                lrow[r] = lrow[r] * alphar[r] + rs;
            }
            // ---- P (C layout) -> per-wave LDS in A-friendly [qrow][s] ----
            _Float16* pW = &sP[wid * 16 * 32];
#pragma unroll
            for (int sub = 0; sub < 2; ++sub)
#pragma unroll
                for (int r = 0; r < 8; ++r)
                    pW[(rb + r) * 32 + sub * 16 + cr] = (_Float16)sf[sub].e[r];
            AFrag pa;
            {
                const _Float16* pp = &pW[ar * 32];
                pa.h[0] = *(const h8v*)&pp[ah];
                pa.h[1] = *(const h8v*)&pp[16 + ah];
            }
            // ---- O = O*alpha + P @ V ----
#pragma unroll
            for (int j = 0; j < 4; ++j) {
#pragma unroll
                for (int r = 0; r < 8; ++r) o[j].e[r] *= alphar[r];
                AFrag bv;
                const _Float16* bp = &sVt[(j * 16 + bc) * 32 + bk2];
                bv.h[0] = *(const h8v*)&bp[0];
                bv.h[1] = *(const h8v*)&bp[16];
                o[j].v = __builtin_amdgcn_wmma_f32_16x16x32_f16(
                    false, pa.v, false, bv.v, (short)0, o[j].v, false, false);
            }
        }
        __syncthreads();
    }

    // normalize and write y (f16, feeds Wo GEMM)
#pragma unroll
    for (int r = 0; r < 8; ++r) {
        const float inv = 1.f / lrow[r];
        const int qrow = qbase + rb + r;
#pragma unroll
        for (int j = 0; j < 4; ++j)
            y[((size_t)(b * T_ + qrow) * HQ_ + hq) * D_ + j * 16 + cr] =
                (_Float16)(o[j].e[r] * inv);
    }
}

// ---------------------------------------------------------------------------
// Router logits: one block per token, one wave per expert (8 waves)
// ---------------------------------------------------------------------------
__global__ __launch_bounds__(256)
void router_kernel(const _Float16* __restrict__ xn, const float* __restrict__ Wr,
                   float* __restrict__ logits)
{
    const int n = blockIdx.x;
    const int lane = threadIdx.x & 31, e = threadIdx.x >> 5;
    const _Float16* xr = xn + (size_t)n * C_;
    float s = 0.f;
    for (int c = lane; c < C_; c += 32) s += (float)xr[c] * Wr[(size_t)c * E_ + e];
#pragma unroll
    for (int o = 16; o > 0; o >>= 1) s += __shfl_xor(s, o, 32);
    if (lane == 0) logits[(size_t)n * E_ + e] = s;
}

__global__ void zero_kernel(float* __restrict__ p, int n)
{
    int i = blockIdx.x * blockDim.x + threadIdx.x;
    if (i < n) p[i] = 0.f;
}

// softmax over E, top-2, stats for aux/z losses; also zeroes combine weights
__global__ void route_topk_kernel(const float* __restrict__ logits,
                                  float* __restrict__ top2v, int* __restrict__ top2i,
                                  float* __restrict__ wcomb,
                                  float* __restrict__ me, float* __restrict__ ce,
                                  float* __restrict__ zacc)
{
    int n = blockIdx.x * blockDim.x + threadIdx.x;
    if (n >= N_) return;
    float l[E_], zsq = 0.f, mx = -3.0e38f;
#pragma unroll
    for (int e = 0; e < E_; ++e) {
        l[e] = logits[(size_t)n * E_ + e];
        zsq += l[e] * l[e];
        mx = fmaxf(mx, l[e]);
    }
    float den = 0.f;
#pragma unroll
    for (int e = 0; e < E_; ++e) { l[e] = __expf(l[e] - mx); den += l[e]; }
    float g[E_];
#pragma unroll
    for (int e = 0; e < E_; ++e) {
        g[e] = l[e] / den;
        wcomb[(size_t)n * E_ + e] = 0.f;
    }
    int i0 = 0;
    for (int e = 1; e < E_; ++e) if (g[e] > g[i0]) i0 = e;
    int i1 = -1;
    for (int e = 0; e < E_; ++e) if (e != i0 && (i1 < 0 || g[e] > g[i1])) i1 = e;
    top2i[n * 2 + 0] = i0;  top2i[n * 2 + 1] = i1;
    top2v[n * 2 + 0] = g[i0]; top2v[n * 2 + 1] = g[i1];
#pragma unroll
    for (int e = 0; e < E_; ++e) atomicAdd(&me[e], g[e] * (1.f / N_));
    atomicAdd(&ce[i0], 1.f / N_);
    atomicAdd(zacc, zsq * (1.f / ((float)N_ * E_)));
}

// FCFS capacity dispatch (token order), matching the reference cumsum rank.
__global__ void dispatch_kernel(const float* __restrict__ top2v, const int* __restrict__ top2i,
                                float* __restrict__ wcomb,
                                const float* __restrict__ me, const float* __restrict__ ce,
                                const float* __restrict__ zacc, float* __restrict__ outAux)
{
    if (threadIdx.x != 0 || blockIdx.x != 0) return;
    int cnt[2][E_];
    for (int k = 0; k < 2; ++k)
        for (int e = 0; e < E_; ++e) cnt[k][e] = 0;
    for (int n = 0; n < N_; ++n)
        for (int k = 0; k < 2; ++k) {
            int e = top2i[n * 2 + k];
            int r = cnt[k][e]++;
            if (r < CAP_) wcomb[(size_t)n * E_ + e] += top2v[n * 2 + k];
        }
    float aux = 0.f;
    for (int e = 0; e < E_; ++e) aux += me[e] * ce[e];
    outAux[0] = (float)E_ * aux;   // aux_loss
    outAux[1] = zacc[0];           // z_loss
}

// act = silu(g) * u, written as f16 (feeds down-proj GEMM)
__global__ void swiglu_kernel(const float* __restrict__ g, const float* __restrict__ u,
                              _Float16* __restrict__ act, unsigned long long n)
{
    unsigned long long i = (unsigned long long)blockIdx.x * blockDim.x + threadIdx.x;
    if (i >= n) return;
    float x = g[i];
    act[i] = (_Float16)((x / (1.f + __expf(-x))) * u[i]);
}

__global__ void copy_kernel(const float* __restrict__ src, float* __restrict__ dst,
                            unsigned long long n)
{
    unsigned long long i = (unsigned long long)blockIdx.x * blockDim.x + threadIdx.x;
    if (i < n) dst[i] = src[i];
}

// ---------------------------------------------------------------------------
extern "C" void kernel_launch(void* const* d_in, const int* in_sizes, int n_in,
                              void* d_out, int out_size, void* d_ws, size_t ws_size,
                              hipStream_t stream)
{
    (void)in_sizes; (void)n_in; (void)out_size; (void)ws_size;
    const float* x    = (const float*)d_in[0];
    const float* rope = (const float*)d_in[1];
    const float* ln1w = (const float*)d_in[2];
    const float* ln2w = (const float*)d_in[3];
    const float* Wq   = (const float*)d_in[4];
    const float* Wk   = (const float*)d_in[5];
    const float* Wv   = (const float*)d_in[6];
    const float* Wo   = (const float*)d_in[7];
    const float* Wr   = (const float*)d_in[8];
    const float* Weg  = (const float*)d_in[9];
    const float* Weu  = (const float*)d_in[10];
    const float* Wed  = (const float*)d_in[11];
    float* out = (float*)d_out;

    // ---- workspace carving (float units) ----
    float* ws = (float*)d_ws;
    size_t off = 0;
    float* qb    = ws + off; off += (size_t)N_ * HQ_  * D_;
    float* kb    = ws + off; off += (size_t)N_ * HKV_ * D_;
    float* vb    = ws + off; off += (size_t)N_ * HKV_ * D_;
    float* hb    = ws + off; off += (size_t)N_ * C_;
    float* gb    = ws + off; off += (size_t)N_ * H_;
    float* ub    = ws + off; off += (size_t)N_ * H_;
    float* logit = ws + off; off += (size_t)N_ * E_;
    float* top2v = ws + off; off += (size_t)N_ * 2;
    float* wcomb = ws + off; off += (size_t)N_ * E_;
    float* me    = ws + off; off += E_;
    float* ce    = ws + off; off += E_;
    float* za    = ws + off; off += 3;
    int*   top2i = (int*)(ws + off); off += (size_t)N_ * 2;
    _Float16* xn1h = (_Float16*)(ws + off); off += (size_t)N_ * C_ / 2;
    _Float16* ybh  = (_Float16*)(ws + off); off += (size_t)N_ * C_ / 2;
    _Float16* xn2h = (_Float16*)(ws + off); off += (size_t)N_ * C_ / 2;
    _Float16* acth = (_Float16*)(ws + off); off += (size_t)N_ * H_ / 2;
    _Float16* WqT  = (_Float16*)(ws + off); off += (size_t)C_ * C_ / 2;
    _Float16* WkT  = (_Float16*)(ws + off); off += (size_t)C_ * HKV_ * D_ / 2;
    _Float16* WvT  = (_Float16*)(ws + off); off += (size_t)C_ * HKV_ * D_ / 2;
    _Float16* WoT  = (_Float16*)(ws + off); off += (size_t)C_ * C_ / 2;
    _Float16* WegT = (_Float16*)(ws + off); off += (size_t)E_ * C_ * H_ / 2;
    _Float16* WeuT = (_Float16*)(ws + off); off += (size_t)E_ * C_ * H_ / 2;
    _Float16* WedT = (_Float16*)(ws + off); off += (size_t)E_ * C_ * H_ / 2;

    // ---- 0) weight transpose+convert to f16 (Bt layout [N][K]) ----
    dim3 tb(32, 8);
    transpose_f32_to_f16<<<dim3(C_ / 32, C_ / 32), tb, 0, stream>>>(Wq, WqT, C_, C_);
    transpose_f32_to_f16<<<dim3((HKV_ * D_) / 32, C_ / 32), tb, 0, stream>>>(Wk, WkT, C_, HKV_ * D_);
    transpose_f32_to_f16<<<dim3((HKV_ * D_) / 32, C_ / 32), tb, 0, stream>>>(Wv, WvT, C_, HKV_ * D_);
    transpose_f32_to_f16<<<dim3(C_ / 32, C_ / 32), tb, 0, stream>>>(Wo, WoT, C_, C_);
    for (int e = 0; e < E_; ++e) {
        transpose_f32_to_f16<<<dim3(H_ / 32, C_ / 32), tb, 0, stream>>>(
            Weg + (size_t)e * C_ * H_, WegT + (size_t)e * C_ * H_, C_, H_);
        transpose_f32_to_f16<<<dim3(H_ / 32, C_ / 32), tb, 0, stream>>>(
            Weu + (size_t)e * C_ * H_, WeuT + (size_t)e * C_ * H_, C_, H_);
        transpose_f32_to_f16<<<dim3(C_ / 32, H_ / 32), tb, 0, stream>>>(
            Wed + (size_t)e * H_ * C_, WedT + (size_t)e * H_ * C_, H_, C_);
    }

    // ---- 1) ln1 + QKV projections (WMMA) ----
    rmsnorm_f16_kernel<<<N_, 256, 0, stream>>>(x, ln1w, xn1h);
    gemm_wmma_f16<<<dim3(C_ / BN, N_ / BM), 256, 0, stream>>>(
        xn1h, WqT, qb, N_, C_, C_, nullptr, 0, nullptr, 0);
    gemm_wmma_f16<<<dim3((HKV_ * D_) / BN, N_ / BM), 256, 0, stream>>>(
        xn1h, WkT, kb, N_, HKV_ * D_, C_, nullptr, 0, nullptr, 0);
    gemm_wmma_f16<<<dim3((HKV_ * D_) / BN, N_ / BM), 256, 0, stream>>>(
        xn1h, WvT, vb, N_, HKV_ * D_, C_, nullptr, 0, nullptr, 0);

    // ---- 2) RoPE + WMMA flash attention + output proj (h = x + y@Wo) ----
    rope_kernel<<<(N_ * HQ_ * (D_ / 2) + 255) / 256, 256, 0, stream>>>(qb, rope, HQ_);
    rope_kernel<<<(N_ * HKV_ * (D_ / 2) + 255) / 256, 256, 0, stream>>>(kb, rope, HKV_);
    flash_attn_wmma<<<dim3(T_ / 64, B_ * HQ_), 128, 0, stream>>>(qb, kb, vb, ybh);
    gemm_wmma_f16<<<dim3(C_ / BN, N_ / BM), 256, 0, stream>>>(
        ybh, WoT, hb, N_, C_, C_, nullptr, 0, x, 0);

    // ---- 3) ln2 + router + top-2 capacity dispatch + losses ----
    rmsnorm_f16_kernel<<<N_, 256, 0, stream>>>(hb, ln2w, xn2h);
    router_kernel<<<N_, 256, 0, stream>>>(xn2h, Wr, logit);
    zero_kernel<<<1, 32, 0, stream>>>(me, 2 * E_ + 1);
    route_topk_kernel<<<(N_ + 255) / 256, 256, 0, stream>>>(
        logit, top2v, top2i, wcomb, me, ce, za);
    dispatch_kernel<<<1, 1, 0, stream>>>(
        top2v, top2i, wcomb, me, ce, za, out + (size_t)N_ * C_);

    // ---- 4) out = h, then accumulate weighted expert outputs ----
    copy_kernel<<<(unsigned)(((size_t)N_ * C_ + 255) / 256), 256, 0, stream>>>(
        hb, out, (unsigned long long)N_ * C_);
    for (int e = 0; e < E_; ++e) {
        gemm_wmma_f16<<<dim3(H_ / BN, N_ / BM), 256, 0, stream>>>(
            xn2h, WegT + (size_t)e * C_ * H_, gb, N_, H_, C_, nullptr, 0, nullptr, 0);
        gemm_wmma_f16<<<dim3(H_ / BN, N_ / BM), 256, 0, stream>>>(
            xn2h, WeuT + (size_t)e * C_ * H_, ub, N_, H_, C_, nullptr, 0, nullptr, 0);
        swiglu_kernel<<<(unsigned)(((size_t)N_ * H_ + 255) / 256), 256, 0, stream>>>(
            gb, ub, acth, (unsigned long long)N_ * H_);
        gemm_wmma_f16<<<dim3(C_ / BN, N_ / BM), 256, 0, stream>>>(
            acth, WedT + (size_t)e * H_ * C_, out, N_, C_, H_, wcomb + e, E_, nullptr, 1);
    }
}
